// GCN_68719476736448
// MI455X (gfx1250) — matrix operand order repaired and checked by reading, not compiled, not run on hardware
//
#include <hip/hip_runtime.h>
#include <math.h>

// ---------------------------------------------------------------------------
// Types for CDNA5 WMMA (wave32): V_WMMA_F32_16X16X32_BF16
// ---------------------------------------------------------------------------
typedef __bf16 v16bf __attribute__((ext_vector_type(16)));
typedef __bf16 v8bf  __attribute__((ext_vector_type(8)));
typedef float  v8f   __attribute__((ext_vector_type(8)));

__device__ __forceinline__ unsigned short f32_to_bf16(float f) {
  union { float f; unsigned u; } c; c.f = f;
  unsigned u = c.u;
  u += 0x7FFFu + ((u >> 16) & 1u);      // round-to-nearest-even
  return (unsigned short)(u >> 16);
}

// ---------------------------------------------------------------------------
// CSR construction: count (int) -> hierarchical scan -> bucket fill.
// cnt starts at 1 (self loop).  deg == cnt, dis = rsqrt(deg).
// ---------------------------------------------------------------------------
__global__ void init_cnt_kernel(int* cnt, int n) {
  int i = blockIdx.x * blockDim.x + threadIdx.x;
  if (i < n) cnt[i] = 1;                // self loop
}

__global__ void count_kernel(const int* __restrict__ dst, int* __restrict__ cnt, int e) {
  int i = blockIdx.x * blockDim.x + threadIdx.x;
  if (i < e) atomicAdd(&cnt[dst[i]], 1);
}

__global__ void dis_kernel(const int* __restrict__ cnt, float* __restrict__ dis, int n) {
  int i = blockIdx.x * blockDim.x + threadIdx.x;
  if (i < n) dis[i] = rsqrtf((float)cnt[i]);   // cnt >= 1 always
}

// Per-block inclusive scan (Hillis-Steele in LDS) + block sums
__global__ void __launch_bounds__(256)
scan1_kernel(const int* __restrict__ cnt, int* __restrict__ partial,
             int* __restrict__ blockSums, int n) {
  __shared__ int sm[256];
  const int t = threadIdx.x;
  const int i = blockIdx.x * 256 + t;
  sm[t] = (i < n) ? cnt[i] : 0;
  __syncthreads();
#pragma unroll
  for (int off = 1; off < 256; off <<= 1) {
    int x = (t >= off) ? sm[t - off] : 0;
    __syncthreads();
    sm[t] += x;
    __syncthreads();
  }
  if (i < n) partial[i] = sm[t];
  if (t == 255) blockSums[blockIdx.x] = sm[255];
}

// Exclusive scan of block sums (single block; nb <= 256)
__global__ void __launch_bounds__(256)
scan2_kernel(const int* __restrict__ blockSums, int* __restrict__ blockOff, int nb) {
  __shared__ int sm[256];
  const int t = threadIdx.x;
  int v = (t < nb) ? blockSums[t] : 0;
  sm[t] = v;
  __syncthreads();
#pragma unroll
  for (int off = 1; off < 256; off <<= 1) {
    int x = (t >= off) ? sm[t - off] : 0;
    __syncthreads();
    sm[t] += x;
    __syncthreads();
  }
  blockOff[t] = sm[t] - v;              // exclusive
}

// rowptr[i+1] = partial[i] + blockOff[i/256]; rowptr[0] = 0
__global__ void scan3_kernel(const int* __restrict__ partial, const int* __restrict__ blockOff,
                             int* __restrict__ rowptr, int n) {
  int i = blockIdx.x * blockDim.x + threadIdx.x;
  if (i < n) rowptr[i + 1] = partial[i] + blockOff[i >> 8];
  if (i == 0) rowptr[0] = 0;
}

__global__ void cursor_kernel(const int* __restrict__ rowptr, int* __restrict__ cursor, int n) {
  int i = blockIdx.x * blockDim.x + threadIdx.x;
  if (i < n) cursor[i] = rowptr[i];
}

// Fill CSR entries: packed (src, norm) per incoming edge (+ self loops)
__global__ void fill_kernel(const int* __restrict__ src, const int* __restrict__ dst,
                            const float* __restrict__ dis, int* __restrict__ cursor,
                            int2* __restrict__ ent, int e, int n) {
  int i = blockIdx.x * blockDim.x + threadIdx.x;
  if (i >= e + n) return;
  int s, d; float nrm;
  if (i < e) { s = src[i]; d = dst[i]; nrm = dis[s] * dis[d]; }
  else       { s = d = i - e; float w = dis[s]; nrm = w * w; }
  int p = atomicAdd(&cursor[d], 1);
  ent[p] = make_int2(s, __float_as_int(nrm));
}

// ---------------------------------------------------------------------------
// fp32 -> bf16 activation conversion (layer-0 input)
// ---------------------------------------------------------------------------
__global__ void convert_kernel(const float* __restrict__ src, unsigned short* __restrict__ dst, int n) {
  int i = blockIdx.x * blockDim.x + threadIdx.x;
  if (i < n) dst[i] = f32_to_bf16(src[i]);
}

// W[fi][fo] fp32 -> Wt[fo][fi] bf16 so B-panel rows are contiguous
__global__ void transpose_w_kernel(const float* __restrict__ W, unsigned short* __restrict__ Wt,
                                   int fi, int fo) {
  int i = blockIdx.x * blockDim.x + threadIdx.x;
  if (i < fi * fo) {
    int r = i / fo, c = i % fo;
    Wt[c * fi + r] = f32_to_bf16(W[i]);
  }
}

// ---------------------------------------------------------------------------
// Z[M,fo] = Hb[M,fi](bf16) @ Wt[fo,fi]^T(bf16), fp32 accumulate, via WMMA.
// Block = 8 waves sharing ONE 64-col B panel staged in LDS via CDNA5 async
// copy (global_load_async_to_lds_b128, ASYNCcnt).  Each wave owns a 16-row
// tile and computes a 16x64 strip (4 accumulators), A streamed from global,
// B served from LDS (ds_load_b128).  Requires fi % 32 == 0, fi <= 512,
// fo % 64 == 0, M % 16 == 0 (true here: 512 / 512|256 / 50000).
// ---------------------------------------------------------------------------
__global__ void __launch_bounds__(256)
gemm_wmma_kernel(const unsigned short* __restrict__ Hb, const unsigned short* __restrict__ Wt,
                 float* __restrict__ Z, int M, int fi, int fo) {
  __shared__ __align__(128) unsigned short smemB[64 * 512];   // 64KB B panel

  const int lane   = threadIdx.x & 31;
  const int wave   = threadIdx.x >> 5;
  const int ntile4 = fo >> 6;                   // 64-col groups
  const int tn     = blockIdx.x % ntile4;
  const int rowBlk = blockIdx.x / ntile4;
  const int colBase = tn * 64;
  const int Mtiles  = M >> 4;

  // ---- cooperative async copy of the 64 x fi B panel into LDS ----
  {
    const char* gpanel = (const char*)(Wt + (size_t)colBase * fi);
    const int nchunk = (64 * fi * 2) >> 4;      // 16B chunks (4096 for fi=512)
    for (int c = threadIdx.x; c < nchunk; c += 256) {
      unsigned lds = (unsigned)(size_t)((const char*)smemB + c * 16);
      unsigned long long ga = (unsigned long long)(size_t)(gpanel + c * 16);
      asm volatile("global_load_async_to_lds_b128 %0, %1, off"
                   :: "v"(lds), "v"(ga) : "memory");
    }
    asm volatile("s_wait_asynccnt 0x0" ::: "memory");
  }
  __syncthreads();

  const int tm = rowBlk * 8 + wave;             // wave-uniform tail guard
  if (tm >= Mtiles) return;

  const int sub  = lane & 15;                   // row for A, col for B/D
  const int half = lane >> 4;                   // K-half selector

  v8f acc[4] = {};
  const unsigned short* aRow = Hb + (size_t)(tm * 16 + sub) * fi;

  for (int k0 = 0; k0 < fi; k0 += 32) {
    // A fragment (16x32 bf16): two contiguous 16B global loads per lane
    v8bf lo = *(const v8bf*)(const void*)(aRow + k0 + half * 16);
    v8bf hi = *(const v8bf*)(const void*)(aRow + k0 + half * 16 + 8);
    v16bf a;
#pragma unroll
    for (int i = 0; i < 8; ++i) { a[i] = lo[i]; a[i + 8] = hi[i]; }

#pragma unroll
    for (int t = 0; t < 4; ++t) {
      // B fragment (32x16 bf16): 32B from the LDS panel
      const unsigned short* bp = smemB + (size_t)(t * 16 + sub) * fi + k0 + half * 16;
      v16bf b = *(const v16bf*)(const void*)bp;
      acc[t] = __builtin_amdgcn_wmma_f32_16x16x32_bf16(
          false, a, false, b, (short)0, acc[t], false, false);
    }
  }

#pragma unroll
  for (int t = 0; t < 4; ++t) {
    float* zp = Z + (size_t)(tm * 16 + half * 8) * fo + colBase + t * 16 + sub;
#pragma unroll
    for (int v = 0; v < 8; ++v) zp[(size_t)v * fo] = acc[t][v];   // D row = v + 8*half
  }
}

// ---------------------------------------------------------------------------
// CSR gather aggregation, one wave32 per node:
//   out[n] = b + sum_{p in rowptr[n]..rowptr[n+1]} norm[p] * Z[src[p]]
// Lane covers 4 float4 chunks (fo=512) / 2 (fo=256), pure register accum —
// no float atomics.  Z (<=102MB) is L2-resident on MI455X (192MB L2).
// Fused epilogue: relu+bf16 (hidden layers) or f32 store (last layer).
// ---------------------------------------------------------------------------
__global__ void __launch_bounds__(256)
gather_kernel(const float* __restrict__ Z, const int2* __restrict__ ent,
              const int* __restrict__ rowptr, const float* __restrict__ bias,
              float* __restrict__ outF, unsigned short* __restrict__ outB,
              int n, int fo, int relu_bf16) {
  const int node = blockIdx.x * 8 + (threadIdx.x >> 5);
  if (node >= n) return;
  const int lane = threadIdx.x & 31;
  const int nj   = fo >> 7;                     // float4 groups per lane (4 or 2)
  const int ch   = fo >> 2;                     // float4 per row

  float4 acc[4];
  const float4* b4 = (const float4*)bias;
#pragma unroll 4
  for (int j = 0; j < nj; ++j) acc[j] = b4[lane + 32 * j];

  const int beg = __builtin_amdgcn_readfirstlane(rowptr[node]);
  const int end = __builtin_amdgcn_readfirstlane(rowptr[node + 1]);
  const float4* Z4 = (const float4*)Z;

  for (int p = beg; p < end; ++p) {
    const int2 epk = ent[p];                    // wave-uniform address
    const float w  = __int_as_float(epk.y);
    const float4* zp = Z4 + (size_t)epk.x * ch;
#pragma unroll 4
    for (int j = 0; j < nj; ++j) {
      const float4 z = zp[lane + 32 * j];
      acc[j].x = fmaf(w, z.x, acc[j].x);
      acc[j].y = fmaf(w, z.y, acc[j].y);
      acc[j].z = fmaf(w, z.z, acc[j].z);
      acc[j].w = fmaf(w, z.w, acc[j].w);
    }
  }

  if (relu_bf16) {
    ushort4* o = (ushort4*)(outB + (size_t)node * fo);
#pragma unroll 4
    for (int j = 0; j < nj; ++j) {
      ushort4 h;
      h.x = f32_to_bf16(fmaxf(acc[j].x, 0.0f));
      h.y = f32_to_bf16(fmaxf(acc[j].y, 0.0f));
      h.z = f32_to_bf16(fmaxf(acc[j].z, 0.0f));
      h.w = f32_to_bf16(fmaxf(acc[j].w, 0.0f));
      o[lane + 32 * j] = h;
    }
  } else {
    float4* o = (float4*)(outF + (size_t)node * fo);
#pragma unroll 4
    for (int j = 0; j < nj; ++j) o[lane + 32 * j] = acc[j];
  }
}

// ---------------------------------------------------------------------------
// log_softmax over C=256: one wave32 per row, 8 values per lane.
// ---------------------------------------------------------------------------
__global__ void __launch_bounds__(256)
logsoftmax_kernel(const float* __restrict__ in, float* __restrict__ out, int n, int C) {
  const int lane = threadIdx.x & 31;
  const int row  = blockIdx.x * 8 + (threadIdx.x >> 5);
  if (row >= n) return;
  const float4* p4 = (const float4*)(in + (size_t)row * C);
  float4 A = p4[lane];
  float4 B = p4[lane + 32];
  float v[8] = {A.x, A.y, A.z, A.w, B.x, B.y, B.z, B.w};

  float m = v[0];
#pragma unroll
  for (int i = 1; i < 8; ++i) m = fmaxf(m, v[i]);
#pragma unroll
  for (int off = 16; off > 0; off >>= 1) m = fmaxf(m, __shfl_xor(m, off, 32));

  float s = 0.0f;
#pragma unroll
  for (int i = 0; i < 8; ++i) s += __expf(v[i] - m);
#pragma unroll
  for (int off = 16; off > 0; off >>= 1) s += __shfl_xor(s, off, 32);

  const float lse = m + __logf(s);
  float4* o4 = (float4*)(out + (size_t)row * C);
  o4[lane]      = make_float4(v[0] - lse, v[1] - lse, v[2] - lse, v[3] - lse);
  o4[lane + 32] = make_float4(v[4] - lse, v[5] - lse, v[6] - lse, v[7] - lse);
}

// ---------------------------------------------------------------------------
// Launcher
// ---------------------------------------------------------------------------
extern "C" void kernel_launch(void* const* d_in, const int* in_sizes, int n_in,
                              void* d_out, int out_size, void* d_ws, size_t ws_size,
                              hipStream_t stream) {
  const float* x  = (const float*)d_in[0];
  const float* Wm[3] = {(const float*)d_in[1], (const float*)d_in[3], (const float*)d_in[5]};
  const float* bm[3] = {(const float*)d_in[2], (const float*)d_in[4], (const float*)d_in[6]};
  const int*   ei = (const int*)d_in[7];

  const int fo0 = in_sizes[2];                  // 512
  const int fi0 = in_sizes[1] / fo0;            // 512
  const int N   = in_sizes[0] / fi0;            // 50000
  const int E   = in_sizes[7] / 2;              // 800000
  const int fdim[4] = {fi0, in_sizes[2], in_sizes[4], in_sizes[6]};  // 512,512,512,256
  const int fmax = fi0;                         // 512

  const int* srcp = ei;
  const int* dstp = ei + E;
  float* out = (float*)d_out;

  // workspace carve (256B aligned)
  char* ws = (char*)d_ws;
  size_t off = 0;
  auto carve = [&](size_t bytes) -> void* {
    off = (off + 255) & ~(size_t)255;
    void* p = ws + off;
    off += bytes;
    return p;
  };
  unsigned short* Hb   = (unsigned short*)carve((size_t)N * fmax * 2);  // bf16 acts
  float*          Zb   = (float*)carve((size_t)N * fmax * 4);           // GEMM out
  float*          Cb   = (float*)carve((size_t)N * fmax * 4);           // agg out (last layer)
  unsigned short* Wt   = (unsigned short*)carve((size_t)fmax * fmax * 2);
  float*          dis  = (float*)carve((size_t)N * 4);
  int*            cnt  = (int*)carve((size_t)N * 4);
  int*            part = (int*)carve((size_t)N * 4);
  int*            bsum = (int*)carve(256 * 4);
  int*            boff = (int*)carve(256 * 4);
  int*            rowp = (int*)carve((size_t)(N + 1) * 4);
  int*            curs = (int*)carve((size_t)N * 4);
  int2*           ent  = (int2*)carve((size_t)(E + N) * 8);
  (void)ws_size; (void)n_in; (void)out_size;

  const int T = 256;
  auto blk = [](long long n, int t) { return (int)((n + t - 1) / t); };
  const int nb = blk(N, 256);                   // scan blocks (<=256)

  // ---- CSR + normalization (shared across all layers) ----
  init_cnt_kernel<<<blk(N, T), T, 0, stream>>>(cnt, N);
  count_kernel<<<blk(E, T), T, 0, stream>>>(dstp, cnt, E);
  dis_kernel<<<blk(N, T), T, 0, stream>>>(cnt, dis, N);
  scan1_kernel<<<nb, 256, 0, stream>>>(cnt, part, bsum, N);
  scan2_kernel<<<1, 256, 0, stream>>>(bsum, boff, nb);
  scan3_kernel<<<blk(N, T), T, 0, stream>>>(part, boff, rowp, N);
  cursor_kernel<<<blk(N, T), T, 0, stream>>>(rowp, curs, N);
  fill_kernel<<<blk((long long)E + N, T), T, 0, stream>>>(srcp, dstp, dis, curs, ent, E, N);

  // ---- x -> bf16 ----
  convert_kernel<<<blk((long long)N * fdim[0], T), T, 0, stream>>>(x, Hb, N * fdim[0]);

  // ---- 3 GCN layers ----
  for (int L = 0; L < 3; ++L) {
    const int fi = fdim[L], fo = fdim[L + 1];
    transpose_w_kernel<<<blk((long long)fi * fo, T), T, 0, stream>>>(Wm[L], Wt, fi, fo);

    const int Mtiles  = N / 16;
    const int rowBlks = (Mtiles + 7) / 8;
    gemm_wmma_kernel<<<rowBlks * (fo / 64), 256, 0, stream>>>(Hb, Wt, Zb, N, fi, fo);

    gather_kernel<<<blk(N, 8), 256, 0, stream>>>(Zb, ent, rowp, bm[L],
                                                 Cb, Hb, N, fo, (L < 2) ? 1 : 0);
  }

  logsoftmax_kernel<<<blk(N, 8), 256, 0, stream>>>(Cb, out, N, fdim[3]);
}
